// GRU_55508157334226
// MI455X (gfx1250) — compile-verified, matrix operand order checked by reading, not run on hardware
//
#include <hip/hip_runtime.h>

// CDNA5 / gfx1250 fused GRU, round 2:
//  - native f32->bf16 casts (v_cvt_pk_bf16_f32) instead of manual RNE bit math
//  - A/B operands gathered as contiguous 8-float runs -> b128 loads
//  - x-side WMMAs issued before the h LDS round-trip; h-side WMMAs last
//  - v_tanh_f32 (guarded) for gate nonlinearity

typedef __attribute__((ext_vector_type(16))) __bf16 v16bf;
typedef __attribute__((ext_vector_type(8)))  float  v8f;

__device__ __forceinline__ __bf16 f2bf(float f) { return (__bf16)f; }

#if __has_builtin(__builtin_amdgcn_tanhf)
__device__ __forceinline__ float fast_tanh(float x) { return __builtin_amdgcn_tanhf(x); }
__device__ __forceinline__ float fast_sigmoid(float x) {
    return __builtin_fmaf(0.5f, __builtin_amdgcn_tanhf(0.5f * x), 0.5f);
}
#else
__device__ __forceinline__ float fast_tanh(float x) {
    return 1.0f - 2.0f * __builtin_amdgcn_rcpf(1.0f + __expf(2.0f * x));
}
__device__ __forceinline__ float fast_sigmoid(float x) {
    return __builtin_amdgcn_rcpf(1.0f + __expf(-x));
}
#endif

__device__ __forceinline__ void put4(v16bf& v, int base, float4 f) {
    v[base + 0] = f2bf(f.x);
    v[base + 1] = f2bf(f.y);
    v[base + 2] = f2bf(f.z);
    v[base + 3] = f2bf(f.w);
}

// Gather one 16-bit 16x32 A/B operand from a row pointer already offset by
// (lane-half * 8): K runs are [0..7] and [16..23] relative to that pointer.
__device__ __forceinline__ v16bf loadA16(const float* __restrict__ p) {
    v16bf v;
    put4(v, 0,  *(const float4*)(p + 0));
    put4(v, 4,  *(const float4*)(p + 4));
    put4(v, 8,  *(const float4*)(p + 16));
    put4(v, 12, *(const float4*)(p + 20));
    return v;
}
__device__ __forceinline__ v16bf loadA16_lds(const float* p) {
    v16bf v;
    put4(v, 0,  *(const float4*)(p + 0));
    put4(v, 4,  *(const float4*)(p + 4));
    put4(v, 8,  *(const float4*)(p + 16));
    put4(v, 12, *(const float4*)(p + 20));
    return v;
}

#define WMMA_BF16(A, B, C) \
    __builtin_amdgcn_wmma_f32_16x16x32_bf16(false, (A), false, (B), (short)0, (C), false, false)

__global__ void __launch_bounds__(32, 1)
gru_fused_kernel(const float* __restrict__ x,
                 const float* __restrict__ W_ih,  // (96,64)
                 const float* __restrict__ W_hh,  // (96,32)
                 const float* __restrict__ b_ih,  // (96)
                 const float* __restrict__ b_hh,  // (96)
                 const float* __restrict__ W_fc,  // (1,32)
                 const float* __restrict__ b_fc,  // (1)
                 float* __restrict__ out)         // (1024,1)
{
    constexpr int T = 512, D = 64, H = 32;
    constexpr int HS = 36;          // padded LDS row stride (floats): conflict-free stores
    const int lane = threadIdx.x;   // 0..31 (wave32)
    const int nl   = lane & 15;
    const int hi   = lane >> 4;     // 0 | 1
    const int b0   = blockIdx.x * 16;

    __shared__ float hbuf[16 * HS]; // h C/D-layout -> A-layout shuffle

    // ---- Preload weights into B-layout bf16 registers (resident all 512 steps) ----
    // B operand: lane holds column N = nl; K element map identical to A map.
    v16bf Wih_B[6][2];
    v16bf Whh_B[6];
    #pragma unroll
    for (int nb = 0; nb < 6; ++nb) {
        const int g = nb * 16 + nl;
        #pragma unroll
        for (int kb = 0; kb < 2; ++kb)
            Wih_B[nb][kb] = loadA16(W_ih + g * D + kb * 32 + hi * 8);
        Whh_B[nb] = loadA16(W_hh + g * H + hi * 8);
    }
    // Per-lane biases (C/D tile column N = nb*16 + nl)
    float bx[6], bh[6];
    #pragma unroll
    for (int nb = 0; nb < 6; ++nb) {
        bx[nb] = b_ih[nb * 16 + nl];
        bh[nb] = b_hh[nb * 16 + nl];
    }

    const v8f vzero = {0.f, 0.f, 0.f, 0.f, 0.f, 0.f, 0.f, 0.f};
    v8f hst[2] = {vzero, vzero};    // h in C/D layout: tile c covers hidden N = c*16..c*16+15

    // Per-lane row pointer for x A-tile gathers (pre-offset by hi*8)
    const float* xrow = x + (size_t)(b0 + nl) * T * D + hi * 8;

    for (int t = 0; t < T; ++t) {
        // ---- h (C/D f32) -> LDS (start the round-trip first) ----
        #pragma unroll
        for (int c = 0; c < 2; ++c)
            #pragma unroll
            for (int r = 0; r < 8; ++r)
                hbuf[(r + 8 * hi) * HS + c * 16 + nl] = hst[c][r];

        // ---- gather x_t tile (16x64 f32) into two 16x32 bf16 A operands ----
        v16bf xA0 = loadA16(xrow);
        v16bf xA1 = loadA16(xrow + 32);
        if (t + 4 < T) {
            __builtin_prefetch(xrow + 4 * D, 0, 0);        // global_prefetch_b8
            __builtin_prefetch(xrow + 4 * D + 32, 0, 0);
        }

        // ---- x-side WMMAs (independent of h round-trip) ----
        v8f acc[6];   // r0,r1,z0,z1,n0,n1 (pre-activation, x part)
        #pragma unroll
        for (int g = 0; g < 6; ++g) {
            acc[g] = WMMA_BF16(xA0, Wih_B[g][0], vzero);
            acc[g] = WMMA_BF16(xA1, Wih_B[g][1], acc[g]);
        }

        // ---- finish h shuffle: LDS -> 16x32 bf16 A operand ----
        __syncthreads();
        v16bf hA = loadA16_lds(&hbuf[nl * HS + hi * 8]);
        __syncthreads();

        // ---- h-side WMMAs ----
        #pragma unroll
        for (int g = 0; g < 4; ++g)                 // r,z: fold into combined pre-act
            acc[g] = WMMA_BF16(hA, Whh_B[g], acc[g]);
        v8f hn0 = WMMA_BF16(hA, Whh_B[4], vzero);   // n-gate h part kept separate
        v8f hn1 = WMMA_BF16(hA, Whh_B[5], vzero);

        // ---- gate elementwise update (layouts of acc/hst tiles coincide) ----
        #pragma unroll
        for (int c = 0; c < 2; ++c) {
            const float br = bx[0 + c] + bh[0 + c];
            const float bz = bx[2 + c] + bh[2 + c];
            const v8f hn = c ? hn1 : hn0;
            #pragma unroll
            for (int r = 0; r < 8; ++r) {
                float rg = fast_sigmoid(acc[0 + c][r] + br);
                float zg = fast_sigmoid(acc[2 + c][r] + bz);
                float ng = fast_tanh(acc[4 + c][r] + bx[4 + c] + rg * (hn[r] + bh[4 + c]));
                hst[c][r] = (1.0f - zg) * ng + zg * hst[c][r];
            }
        }
        xrow += D;
    }

    // ---- final: out = h_last @ W_fc^T + b_fc ----
    #pragma unroll
    for (int c = 0; c < 2; ++c)
        #pragma unroll
        for (int r = 0; r < 8; ++r)
            hbuf[(r + 8 * hi) * HS + c * 16 + nl] = hst[c][r];
    __syncthreads();
    if (lane < 16) {
        float acc = b_fc[0];
        #pragma unroll
        for (int n = 0; n < H; ++n)
            acc += hbuf[lane * HS + n] * W_fc[n];
        out[b0 + lane] = acc;
    }
}

extern "C" void kernel_launch(void* const* d_in, const int* in_sizes, int n_in,
                              void* d_out, int out_size, void* d_ws, size_t ws_size,
                              hipStream_t stream) {
    (void)in_sizes; (void)n_in; (void)d_ws; (void)ws_size; (void)out_size;
    const float* x    = (const float*)d_in[0];
    const float* W_ih = (const float*)d_in[1];
    const float* W_hh = (const float*)d_in[2];
    const float* b_ih = (const float*)d_in[3];
    const float* b_hh = (const float*)d_in[4];
    const float* W_fc = (const float*)d_in[5];
    const float* b_fc = (const float*)d_in[6];
    float* out = (float*)d_out;

    // 1024 batch rows / 16-row WMMA tiles = 64 waves, one wave32 per block
    gru_fused_kernel<<<64, 32, 0, stream>>>(x, W_ih, W_hh, b_ih, b_hh, W_fc, b_fc, out);
}
